// ResBlock_5016521801758
// MI455X (gfx1250) — matrix operand order repaired and checked by reading, not compiled, not run on hardware
//
#include <hip/hip_runtime.h>
#include <hip/hip_bf16.h>

typedef __attribute__((ext_vector_type(16))) _Float16 v16h;
typedef __attribute__((ext_vector_type(8)))  _Float16 v8h;
typedef __attribute__((ext_vector_type(8)))  float    v8f;

#define COUT 128
#define NEG_SLOPE 0.01f
#define BN_EPS 1e-5f
#define WPAD 8  // per-row LDS padding (halves) -> 272B stride, conflict-free b128 reads

#if __has_builtin(__builtin_amdgcn_global_load_async_to_lds_b128)
#define HAVE_ASYNC_LDS 1
typedef int v4i __attribute__((vector_size(16)));
typedef __attribute__((address_space(1))) v4i g_v4i;  // global (prints as __device__)
typedef __attribute__((address_space(3))) v4i l_v4i;  // LDS (prints as __shared__)
#else
#define HAVE_ASYNC_LDS 0
#endif

__device__ __forceinline__ void wait_async_lds() {
#if __has_builtin(__builtin_amdgcn_s_wait_asynccnt)
    __builtin_amdgcn_s_wait_asynccnt(0);
#else
    asm volatile("s_wait_asynccnt 0" ::: "memory");
#endif
}

// ---------------------------------------------------------------------------
// feats f32 [N,64] -> f16 [N+1,64] with zero pad row at index N
__global__ __launch_bounds__(256) void cvt_feats(const float* __restrict__ in,
                                                 _Float16* __restrict__ out,
                                                 int n) {
    long idx = (long)blockIdx.x * blockDim.x + threadIdx.x;
    long total = (long)(n + 1) * 64;
    if (idx >= total) return;
    long v = idx >> 6;
    out[idx] = (v == n) ? (_Float16)0.0f : (_Float16)in[idx];
}

// W [9, CIN, 128] f32 -> Wt [9, 128, CIN] f16   (B-matrix friendly: K contiguous)
__global__ __launch_bounds__(256) void cvt_weight(const float* __restrict__ W,
                                                  _Float16* __restrict__ Wt,
                                                  int cin) {
    long idx = (long)blockIdx.x * blockDim.x + threadIdx.x;
    long total = 9L * COUT * cin;
    if (idx >= total) return;
    long off = idx / ((long)COUT * cin);
    long rem = idx - off * COUT * cin;
    long d = rem / cin;      // cout index
    long c = rem - d * cin;  // cin  index
    Wt[idx] = (_Float16)W[(off * cin + c) * COUT + d];
}

__global__ __launch_bounds__(256) void zero_stats(float* __restrict__ s) {
    if (threadIdx.x < 256) s[threadIdx.x] = 0.0f;  // sum[128] ++ sumsq[128]
}

// ---------------------------------------------------------------------------
// Stage one offset's weight slice [COUT, CIN] into LDS with row padding.
// CHUNKS is a multiple of 256 for both CIN=64 (1024) and CIN=128 (2048):
// no divergence, uniform trip count.
template <int CIN>
__device__ __forceinline__ void stage_weights(const _Float16* __restrict__ Wt,
                                              int off, _Float16* lbuf, int tid) {
    const int CHUNKS = COUT * CIN / 8;  // 16B chunks
    const _Float16* src = Wt + (long)off * COUT * CIN;
#pragma unroll
    for (int it = 0; it < CHUNKS / 256; ++it) {
        int c = it * 256 + tid;
        int rowi = (c * 8) / CIN;
        int col  = (c * 8) % CIN;
        const _Float16* g = src + c * 8;
        _Float16* l = lbuf + rowi * (CIN + WPAD) + col;
#if HAVE_ASYNC_LDS
        __builtin_amdgcn_global_load_async_to_lds_b128((g_v4i*)g, (l_v4i*)l, 0, 0);
#else
        *(v8h*)l = *(const v8h*)g;
#endif
    }
}

// Fused submanifold conv (9 offsets) + LeakyReLU + BN-stat accumulation.
// Wave = 16 voxels x 128 cout.  Block = 8 waves = 128 voxels (tail-guarded).
// Weights double-buffered in LDS; B fragments served by ds_load_b128.
template <int CIN>
__global__ __launch_bounds__(256) void subm_conv_wmma(
    const _Float16* __restrict__ X,   // [N+1, CIN] padded, pad row zero
    const _Float16* __restrict__ Wt,  // [9, 128, CIN]
    const int* __restrict__ nbr,      // [9, N], entry==N -> pad row
    float* __restrict__ Y,            // [N, 128] post-lrelu, pre-BN
    float* __restrict__ gsum,         // [128]
    float* __restrict__ gsumsq,       // [128]
    int n) {
    const int LSTRIDE = CIN + WPAD;
    __shared__ _Float16 wbuf[2][COUT * (CIN + WPAD)];
    __shared__ float ls[COUT];
    __shared__ float lq[COUT];

    int tid = threadIdx.x;
    if (tid < COUT) { ls[tid] = 0.0f; lq[tid] = 0.0f; }

    int wave = tid >> 5;
    int lane = tid & 31;
    int m16  = lane & 15;   // A row within tile / B column within n-tile
    int kh   = lane >> 4;   // lane half
    int row0 = (blockIdx.x * 8 + wave) * 16;
    bool rowActive = (row0 + m16) < n;  // gather guard (tail tiles read pad row)

    v8f c[8];
    v8f zero = {};
#pragma unroll
    for (int i = 0; i < 8; ++i) c[i] = zero;

    stage_weights<CIN>(Wt, 0, wbuf[0], tid);
    wait_async_lds();
    __syncthreads();

    for (int off = 0; off < 9; ++off) {
        if (off < 8) stage_weights<CIN>(Wt, off + 1, wbuf[(off + 1) & 1], tid);

        int vrow = rowActive ? nbr[(long)off * n + row0 + m16] : n;
        const _Float16* xr = X + (long)vrow * CIN;
        const _Float16* wl = wbuf[off & 1];
#pragma unroll
        for (int kt = 0; kt < CIN / 32; ++kt) {
            // A fragment: lanes 0-15 K=kt*32+{0..7,16..23}; lanes 16-31 +8
            union { v16h v; v8h h[2]; } a;
            a.h[0] = *(const v8h*)(xr + kt * 32 + 8 * kh);
            a.h[1] = *(const v8h*)(xr + kt * 32 + 16 + 8 * kh);
#pragma unroll
            for (int nt = 0; nt < 8; ++nt) {
                // B fragment from LDS: col = nt*16+m16, K = kt*32+{0..7,16..23} + 8*kh
                const _Float16* wp = wl + (nt * 16 + m16) * LSTRIDE + kt * 32 + 8 * kh;
                union { v16h v; v8h h[2]; } b;
                b.h[0] = *(const v8h*)(wp);
                b.h[1] = *(const v8h*)(wp + 16);
                c[nt] = __builtin_amdgcn_wmma_f32_16x16x32_f16(
                    false, a.v, false, b.v, (short)0, c[nt], false, false);
            }
        }
        wait_async_lds();
        __syncthreads();
    }

    // Epilogue: LeakyReLU, store Y, per-channel partial stats (tail-guarded).
#pragma unroll
    for (int nt = 0; nt < 8; ++nt) {
        int ncol = nt * 16 + m16;
        float s = 0.0f, q = 0.0f;
#pragma unroll
        for (int r = 0; r < 8; ++r) {
            int mrow = row0 + r + 8 * kh;  // C layout: lanes16-31 -> M=r+8
            float val = c[nt][r];
            val = (val >= 0.0f) ? val : NEG_SLOPE * val;
            if (mrow < n) {
                Y[(long)mrow * COUT + ncol] = val;
                s += val;
                q += val * val;
            }
        }
        atomicAdd(&ls[ncol], s);
        atomicAdd(&lq[ncol], q);
    }
    __syncthreads();
    if (tid < COUT) {
        atomicAdd(&gsum[tid], ls[tid]);
        atomicAdd(&gsumsq[tid], lq[tid]);
    }
}

// ---------------------------------------------------------------------------
__global__ __launch_bounds__(COUT) void bn_finalize(const float* __restrict__ gsum,
                                                    const float* __restrict__ gsumsq,
                                                    const float* __restrict__ gamma,
                                                    const float* __restrict__ beta,
                                                    float* __restrict__ scale,
                                                    float* __restrict__ shift,
                                                    int n) {
    int cidx = threadIdx.x;
    float inv_n = 1.0f / (float)n;
    float m = gsum[cidx] * inv_n;
    float v = gsumsq[cidx] * inv_n - m * m;
    float sc = gamma[cidx] * rsqrtf(v + BN_EPS);
    scale[cidx] = sc;
    shift[cidx] = beta[cidx] - m * sc;
}

// Y f32 [N,128] -> f16 [N+1,128] normalized, zero pad row
__global__ __launch_bounds__(256) void bn_apply_f16(const float* __restrict__ Y,
                                                    const float* __restrict__ scale,
                                                    const float* __restrict__ shift,
                                                    _Float16* __restrict__ out,
                                                    int n) {
    long idx = (long)blockIdx.x * blockDim.x + threadIdx.x;
    long total = (long)(n + 1) * COUT;
    if (idx >= total) return;
    long v = idx >> 7;
    int cidx = (int)(idx & (COUT - 1));
    out[idx] = (v == n) ? (_Float16)0.0f
                        : (_Float16)(Y[idx] * scale[cidx] + shift[cidx]);
}

// shortcut: d_out = bn(Y)  (f32)
__global__ __launch_bounds__(256) void bn_apply_f32(const float* __restrict__ Y,
                                                    const float* __restrict__ scale,
                                                    const float* __restrict__ shift,
                                                    float* __restrict__ out,
                                                    int n) {
    long idx = (long)blockIdx.x * blockDim.x + threadIdx.x;
    if (idx >= (long)n * COUT) return;
    int cidx = (int)(idx & (COUT - 1));
    out[idx] = Y[idx] * scale[cidx] + shift[cidx];
}

// final: d_out += bn(Y)
__global__ __launch_bounds__(256) void bn_apply_add(const float* __restrict__ Y,
                                                    const float* __restrict__ scale,
                                                    const float* __restrict__ shift,
                                                    float* __restrict__ out,
                                                    int n) {
    long idx = (long)blockIdx.x * blockDim.x + threadIdx.x;
    if (idx >= (long)n * COUT) return;
    int cidx = (int)(idx & (COUT - 1));
    out[idx] = out[idx] + Y[idx] * scale[cidx] + shift[cidx];
}

// ---------------------------------------------------------------------------
extern "C" void kernel_launch(void* const* d_in, const int* in_sizes, int n_in,
                              void* d_out, int out_size, void* d_ws, size_t ws_size,
                              hipStream_t stream) {
    const float* feats = (const float*)d_in[0];
    const float* W1    = (const float*)d_in[1];
    const float* W1_2  = (const float*)d_in[2];
    const float* W2    = (const float*)d_in[3];
    const float* W3    = (const float*)d_in[4];
    const float* g0    = (const float*)d_in[5];
    const float* b0    = (const float*)d_in[6];
    const float* g0_2  = (const float*)d_in[7];
    const float* b0_2  = (const float*)d_in[8];
    const float* g1    = (const float*)d_in[9];
    const float* b1    = (const float*)d_in[10];
    const float* g2    = (const float*)d_in[11];
    const float* b2    = (const float*)d_in[12];
    const int* nbr31   = (const int*)d_in[13];
    const int* nbr13   = (const int*)d_in[14];
    float* out = (float*)d_out;

    const int N = in_sizes[0] / 64;  // 200000
    const int CIN = 64;

    // workspace carve-out (256B aligned)
    char* wsp = (char*)d_ws;
    size_t o = 0;
    auto alloc = [&](size_t bytes) {
        void* p = wsp + o;
        o = (o + bytes + 255) & ~(size_t)255;
        return p;
    };
    _Float16* Xf16  = (_Float16*)alloc((size_t)(N + 1) * CIN * 2);
    _Float16* Sf16  = (_Float16*)alloc((size_t)(N + 1) * COUT * 2);
    float*    Y     = (float*)alloc((size_t)N * COUT * 4);
    _Float16* Wt1   = (_Float16*)alloc(9L * COUT * CIN * 2);
    _Float16* Wt2   = (_Float16*)alloc(9L * COUT * CIN * 2);
    _Float16* Wt12  = (_Float16*)alloc(9L * COUT * COUT * 2);
    _Float16* Wt3   = (_Float16*)alloc(9L * COUT * COUT * 2);
    float*    stats = (float*)alloc(256 * 4);  // sum[128] ++ sumsq[128]
    float*    scale = (float*)alloc(COUT * 4);
    float*    shift = (float*)alloc(COUT * 4);
    float* gsum = stats;
    float* gsq  = stats + COUT;

    const int tiles = (N + 15) / 16;
    const int convGrid = (tiles + 7) / 8;  // 8 waves (16 voxels each) per block
    long nElemPad64  = (long)(N + 1) * CIN;
    long nElemPad128 = (long)(N + 1) * COUT;
    long nElem128    = (long)N * COUT;
    int gPad64  = (int)((nElemPad64 + 255) / 256);
    int gPad128 = (int)((nElemPad128 + 255) / 256);
    int g128    = (int)((nElem128 + 255) / 256);
    int gW64    = (int)((9L * COUT * CIN + 255) / 256);
    int gW128   = (int)((9L * COUT * COUT + 255) / 256);

    // prep
    cvt_feats<<<gPad64, 256, 0, stream>>>(feats, Xf16, N);
    cvt_weight<<<gW64, 256, 0, stream>>>(W1, Wt1, CIN);
    cvt_weight<<<gW128, 256, 0, stream>>>(W1_2, Wt12, COUT);
    cvt_weight<<<gW64, 256, 0, stream>>>(W2, Wt2, CIN);
    cvt_weight<<<gW128, 256, 0, stream>>>(W3, Wt3, COUT);

    // shortcut = bn0(lrelu(conv(X, W1, nbr31)))
    zero_stats<<<1, 256, 0, stream>>>(stats);
    subm_conv_wmma<64><<<convGrid, 256, 0, stream>>>(Xf16, Wt1, nbr31, Y, gsum, gsq, N);
    bn_finalize<<<1, COUT, 0, stream>>>(gsum, gsq, g0, b0, scale, shift, N);
    bn_apply_f16<<<gPad128, 256, 0, stream>>>(Y, scale, shift, Sf16, N);

    // shortcut = bn0_2(lrelu(conv(shortcut, W1_2, nbr13))) -> d_out
    zero_stats<<<1, 256, 0, stream>>>(stats);
    subm_conv_wmma<128><<<convGrid, 256, 0, stream>>>(Sf16, Wt12, nbr13, Y, gsum, gsq, N);
    bn_finalize<<<1, COUT, 0, stream>>>(gsum, gsq, g0_2, b0_2, scale, shift, N);
    bn_apply_f32<<<g128, 256, 0, stream>>>(Y, scale, shift, out, N);

    // resA = bn1(lrelu(conv(X, W2, nbr13)))
    zero_stats<<<1, 256, 0, stream>>>(stats);
    subm_conv_wmma<64><<<convGrid, 256, 0, stream>>>(Xf16, Wt2, nbr13, Y, gsum, gsq, N);
    bn_finalize<<<1, COUT, 0, stream>>>(gsum, gsq, g1, b1, scale, shift, N);
    bn_apply_f16<<<gPad128, 256, 0, stream>>>(Y, scale, shift, Sf16, N);

    // resA = bn2(lrelu(conv(resA, W3, nbr31)));  d_out += resA
    zero_stats<<<1, 256, 0, stream>>>(stats);
    subm_conv_wmma<128><<<convGrid, 256, 0, stream>>>(Sf16, Wt3, nbr31, Y, gsum, gsq, N);
    bn_finalize<<<1, COUT, 0, stream>>>(gsum, gsq, g2, b2, scale, shift, N);
    bn_apply_add<<<g128, 256, 0, stream>>>(Y, scale, shift, out, N);

    (void)n_in; (void)out_size; (void)ws_size;
}
